// Edge_18013092840065
// MI455X (gfx1250) — matrix operand order repaired and checked by reading, not compiled, not run on hardware
//
#include <hip/hip_runtime.h>
#include <hip/hip_fp16.h>
#include <math.h>

typedef _Float16 h4  __attribute__((ext_vector_type(4)));
typedef _Float16 h8  __attribute__((ext_vector_type(8)));
typedef _Float16 h16 __attribute__((ext_vector_type(16)));
typedef float    f4v __attribute__((ext_vector_type(4)));
typedef float    f8  __attribute__((ext_vector_type(8)));

#define WAVES       8
#define F           64
#define KMSG        128
#define MSG_STRIDE  136   // 128 + 8 pad (f16); 272B row = 16B multiple, conflict-friendly
#define H_STRIDE    72    // 64 + 8 pad  (f16); 144B row = 16B multiple
#define W1_STRIDE   136
#define W2_STRIDE   72

// D = A(16x32 f16) * B(32x16 f16) + C(16x16 f32)
static __device__ __forceinline__ f8 wmma_f16(h16 a, h16 b, f8 c) {
    return __builtin_amdgcn_wmma_f32_16x16x32_f16(false, a, false, b, (short)0, c, false, false);
}

static __device__ __forceinline__ f8 splat8(float v) {
    return (f8){v, v, v, v, v, v, v, v};
}

__global__ __launch_bounds__(256) void edgeconv_wmma_kernel(
    const float* __restrict__ x,   // [N, 64]
    const int*   __restrict__ ei,  // [2, E]  row0 = src(j), row1 = dst(i)
    const float* __restrict__ W1,  // [128, 64]
    const float* __restrict__ b1,  // [64]
    const float* __restrict__ W2,  // [64, 64]
    const float* __restrict__ b2,  // [64]
    float*       __restrict__ out, // [N, 64] f32 accumulator (pre-zeroed)
    int E, int numTiles, int iters)
{
    __shared__ _Float16 sW1t[F * W1_STRIDE];          // W1^T: [n][k], k=0..127
    __shared__ _Float16 sW2t[F * W2_STRIDE];          // W2^T: [n][k], k=0..63
    __shared__ float    sB1[F];
    __shared__ float    sB2[F];
    __shared__ _Float16 sMsg[WAVES][16 * MSG_STRIDE]; // per-wave [row=edge][k]
    __shared__ _Float16 sH[WAVES][16 * H_STRIDE];     // per-wave [row=edge][k]
    __shared__ int      sDst[WAVES][16];
    __shared__ int      sSrc[WAVES][16];

    // ---- stage weights (f32 -> f16, transposed) ----
    for (int idx = threadIdx.x; idx < KMSG * F; idx += 256) {
        int k = idx >> 6, n = idx & 63;               // coalesced read of W1[k][n]
        sW1t[n * W1_STRIDE + k] = (_Float16)W1[idx];
    }
    for (int idx = threadIdx.x; idx < F * F; idx += 256) {
        int k = idx >> 6, n = idx & 63;
        sW2t[n * W2_STRIDE + k] = (_Float16)W2[idx];
    }
    if (threadIdx.x < F) {
        sB1[threadIdx.x] = b1[threadIdx.x];
        sB2[threadIdx.x] = b2[threadIdx.x];
    }
    __syncthreads();

    const int wave = threadIdx.x >> 5;
    const int lane = threadIdx.x & 31;
    const int step = gridDim.x * WAVES;

    // WMMA fragment geometry (wave32, 16x16x32 f16)
    const int arow  = lane & 15;                 // A: row of tile this lane feeds
    const int akoff = (lane < 16) ? 0 : 16;      // A: contiguous 16 K values
    const int bn    = lane & 15;                 // B: column this lane feeds
    const int bkoff = (lane < 16) ? 0 : 16;      // B: contiguous 16 K values
    const int moff  = (lane < 16) ? 0 : 8;       // C/D: VGPR v -> row moff+v, col = lane&15

    _Float16* msg = &sMsg[wave][0];
    _Float16* hld = &sH[wave][0];

    // hoist per-lane biases (lane-invariant across tiles)
    float bias1[4], bias2[4];
#pragma unroll
    for (int nb = 0; nb < 4; ++nb) {
        bias1[nb] = sB1[nb * 16 + bn];
        bias2[nb] = sB2[nb * 16 + bn];
    }

    for (int it = 0; it < iters; ++it) {
        const int tile  = blockIdx.x * WAVES + wave + it * step;
        const bool valid = (tile < numTiles);
        const int ebase = tile * 16;

        // ---- 1) edge indices into LDS (lane 0..15) ----
        if (valid && lane < 16) {
            int e = ebase + lane;
            int s = 0, d = 0;
            if (e < E) { s = ei[e]; d = ei[E + e]; }
            sSrc[wave][lane] = s;
            sDst[wave][lane] = d;
        }

        // ---- 2) gather x_i / x_j, build f16 msg tile [16 x 128] ----
        if (valid) {
            const int erow  = lane >> 1;
            const int cbase = (lane & 1) * 32;
            const int e     = ebase + erow;
            const unsigned s = (unsigned)sSrc[wave][erow];
            const unsigned d = (unsigned)sDst[wave][erow];
            // 32-bit element offsets -> saddr + voffset global loads
            const unsigned offi = d * (unsigned)F + (unsigned)cbase;
            const unsigned offj = s * (unsigned)F + (unsigned)cbase;
            _Float16* mrow = msg + erow * MSG_STRIDE;
            if (e < E) {
#pragma unroll
                for (int q = 0; q < 8; ++q) {
                    f4v a = *(const f4v*)(x + offi + q * 4);
                    f4v b = *(const f4v*)(x + offj + q * 4);
                    *(h4*)(mrow + cbase + q * 4)      = __builtin_convertvector(a, h4);
                    *(h4*)(mrow + F + cbase + q * 4)  = __builtin_convertvector(b - a, h4);
                }
            } else {
                h4 z = {};
#pragma unroll
                for (int q = 0; q < 8; ++q) {
                    *(h4*)(mrow + cbase + q * 4)     = z;
                    *(h4*)(mrow + F + cbase + q * 4) = z;
                }
            }
        }

        // ---- 3) GEMM1: [16x128] x [128x64] + b1 -> relu -> h ----
        f8 acc[4];
#pragma unroll
        for (int nb = 0; nb < 4; ++nb) acc[nb] = splat8(bias1[nb]);  // bias in C init

        if (valid) {
            const _Float16* mrow = msg + arow * MSG_STRIDE;
#pragma unroll
            for (int kb = 0; kb < 4; ++kb) {
                const int k0 = kb * 32;
                h8 alo = *(const h8*)(mrow + k0 + akoff);
                h8 ahi = *(const h8*)(mrow + k0 + akoff + 8);
                h16 A = __builtin_shufflevector(alo, ahi,
                        0,1,2,3,4,5,6,7,8,9,10,11,12,13,14,15);
#pragma unroll
                for (int nb = 0; nb < 4; ++nb) {
                    const _Float16* wrow = &sW1t[(nb * 16 + bn) * W1_STRIDE + k0 + bkoff];
                    h8 blo = *(const h8*)(wrow);
                    h8 bhi = *(const h8*)(wrow + 8);
                    h16 B = __builtin_shufflevector(blo, bhi,
                            0,1,2,3,4,5,6,7,8,9,10,11,12,13,14,15);
                    acc[nb] = wmma_f16(A, B, acc[nb]);
                }
            }
            // relu, write h tile [16 x 64] f16
#pragma unroll
            for (int nb = 0; nb < 4; ++nb) {
                const int ncol = nb * 16 + bn;
#pragma unroll
                for (int v = 0; v < 8; ++v) {
                    float t = acc[nb][v];
                    t = t > 0.0f ? t : 0.0f;
                    hld[(moff + v) * H_STRIDE + ncol] = (_Float16)t;
                }
            }
        }

        // ---- 4) GEMM2: [16x64] x [64x64] + b2 ----
        f8 acc2[4];
#pragma unroll
        for (int nb = 0; nb < 4; ++nb) acc2[nb] = splat8(bias2[nb]); // bias in C init

        if (valid) {
            const _Float16* hrow = hld + arow * H_STRIDE;
#pragma unroll
            for (int kb = 0; kb < 2; ++kb) {
                const int k0 = kb * 32;
                h8 alo = *(const h8*)(hrow + k0 + akoff);
                h8 ahi = *(const h8*)(hrow + k0 + akoff + 8);
                h16 A = __builtin_shufflevector(alo, ahi,
                        0,1,2,3,4,5,6,7,8,9,10,11,12,13,14,15);
#pragma unroll
                for (int nb = 0; nb < 4; ++nb) {
                    const _Float16* wrow = &sW2t[(nb * 16 + bn) * W2_STRIDE + k0 + bkoff];
                    h8 blo = *(const h8*)(wrow);
                    h8 bhi = *(const h8*)(wrow + 8);
                    h16 B = __builtin_shufflevector(blo, bhi,
                            0,1,2,3,4,5,6,7,8,9,10,11,12,13,14,15);
                    acc2[nb] = wmma_f16(A, B, acc2[nb]);
                }
            }
            // ---- 5) scatter-add: one base address per edge, imm offsets per nb ----
#pragma unroll
            for (int v = 0; v < 8; ++v) {
                const int m = moff + v;
                const int e = ebase + m;
                if (e < E) {
                    const unsigned base = (unsigned)sDst[wave][m] * (unsigned)F + (unsigned)bn;
#pragma unroll
                    for (int nb = 0; nb < 4; ++nb) {
                        atomicAdd(out + base + nb * 16, acc2[nb][v]);
                    }
                }
            }
        }
    }
}

__global__ __launch_bounds__(256) void tanh_finalize_kernel(float* __restrict__ out, int n)
{
    int i = blockIdx.x * blockDim.x + threadIdx.x;
    int stride = gridDim.x * blockDim.x;
    for (; i < n; i += stride) {
        out[i] = tanhf(out[i]);
    }
}

extern "C" void kernel_launch(void* const* d_in, const int* in_sizes, int n_in,
                              void* d_out, int out_size, void* d_ws, size_t ws_size,
                              hipStream_t stream) {
    const float* x  = (const float*)d_in[0];
    const int*   ei = (const int*)  d_in[1];
    const float* W1 = (const float*)d_in[2];
    const float* b1 = (const float*)d_in[3];
    const float* W2 = (const float*)d_in[4];
    const float* b2 = (const float*)d_in[5];
    float* out = (float*)d_out;

    const int E = in_sizes[1] / 2;
    const int numTiles = (E + 15) / 16;

    // zero the f32 accumulator (d_out is poisoned by the harness)
    hipMemsetAsync(d_out, 0, (size_t)out_size * sizeof(float), stream);

    int grid = 1024;
    int needed = (numTiles + WAVES - 1) / WAVES;
    if (grid > needed) grid = needed;
    if (grid < 1) grid = 1;
    const int step = grid * WAVES;
    const int iters = (numTiles + step - 1) / step;

    edgeconv_wmma_kernel<<<grid, 256, 0, stream>>>(x, ei, W1, b1, W2, b2, out,
                                                   E, numTiles, iters);

    int tgrid = (out_size + 255) / 256;
    if (tgrid > 4096) tgrid = 4096;
    tanh_finalize_kernel<<<tgrid, 256, 0, stream>>>(out, out_size);
}
